// RandomSparseNewMlp_47304769798189
// MI455X (gfx1250) — matrix operand order repaired and checked by reading, not compiled, not run on hardware
//
#include <hip/hip_runtime.h>
#include <hip/hip_bf16.h>
#include <math.h>

// ---------- problem dims ----------
#define IN_F   1024
#define HID_F  4096
#define OUT_F  1024
#define BATCH  64
#define TOKS   768
#define M2     (BATCH * TOKS)      // 49152 rows for GEMM2
#define KT1    (HID_F / 32)        // 128 K-steps in GEMM1
#define KT2    (IN_F  / 32)        // 32  K-steps in GEMM2 / #ktiles in swizzled B

// ---------- types ----------
typedef __attribute__((ext_vector_type(16))) __bf16 v16bf;
typedef __attribute__((ext_vector_type(8)))  float  v8f;

struct alignas(16) f4u { float x, y, z, w; };
struct alignas(16) u4  { unsigned x, y, z, w; };
struct alignas(8)  u2  { unsigned x, y; };

union FragU { u4 q[2]; v16bf v; };     // 32B: one 16-elem bf16 WMMA operand
union AccU  { v8f v; float f[8]; };    // 32B: one 16x16 f32 accumulator tile

__device__ __forceinline__ unsigned fbits(float f) {
    union { float f; unsigned u; } v; v.f = f; return v.u;
}

// ---------- f32 -> packed bf16x2 (cheap: round-half-up + byte perm) ----------
#if __has_builtin(__builtin_amdgcn_cvt_pk_bf16_f32)
typedef __attribute__((ext_vector_type(2))) __bf16 v2bf;
__device__ __forceinline__ unsigned pk2(float lo, float hi) {
    union { v2bf v; unsigned u; } c;
    c.v = __builtin_amdgcn_cvt_pk_bf16_f32(lo, hi);
    return c.u;
}
#elif __has_builtin(__builtin_amdgcn_perm)
__device__ __forceinline__ unsigned pk2(float lo, float hi) {
    // v_perm_b32: bytes {hi.b3, hi.b2, lo.b3, lo.b2} after +0x8000 rounding
    return __builtin_amdgcn_perm(fbits(hi) + 0x8000u, fbits(lo) + 0x8000u,
                                 0x07060302u);
}
#else
__device__ __forceinline__ unsigned pk2(float lo, float hi) {
    return ((fbits(hi) + 0x8000u) & 0xffff0000u) | ((fbits(lo) + 0x8000u) >> 16);
}
#endif

__device__ __forceinline__ unsigned f2bf1(float f) {   // scalar f32->bf16
    return (fbits(f) + 0x8000u) >> 16;
}

// =====================================================================
// Kernel 1: attn[s,t] = (sum_d fc1w[d,s]*fc2w[t,d] + b[t]) * mask[s,t]
// Output written as bf16 in WMMA-B-fragment swizzled layout:
//   bsw[(((t/16)*KT2 + (s/32))*32 + lane)*16 + (s&15)], lane=(t&15)+16*((s>>4)&1)
// WG tile: 128(s) x 256(t); wave tile 64x64; K-step 32, double-buffered LDS.
// =====================================================================
__global__ __launch_bounds__(256)
void gemm1_bias_mask(const float* __restrict__ fc1w,   // [HID_F][IN_F]
                     const float* __restrict__ fc2w,   // [OUT_F][HID_F]
                     const float* __restrict__ fc2b,   // [OUT_F]
                     const float* __restrict__ mask,   // [IN_F][OUT_F]
                     unsigned short* __restrict__ bsw) // 1024*1024 bf16
{
    __shared__ alignas(16) unsigned short ldsA[2][128 * 40];     // [s][d] stride 40
    __shared__ alignas(16) unsigned short ldsB[2][16 * 32 * 16]; // fragment layout

    const int tid  = threadIdx.x;
    const int lane = tid & 31;
    const int wave = tid >> 5;
    const int wm   = wave >> 2;      // 0..1  (64 s-rows each)
    const int wn   = wave & 3;       // 0..3  (64 t-cols each)
    const int s0   = blockIdx.y * 128;
    const int t0   = blockIdx.x * 256;

    AccU acc[4][4];
    #pragma unroll
    for (int i = 0; i < 4; ++i)
        #pragma unroll
        for (int j = 0; j < 4; ++j)
            #pragma unroll
            for (int r = 0; r < 8; ++r) acc[i][j].f[r] = 0.0f;

    // A staging: 4(d) x 4(s) micro-block per thread (register transpose)
    const int dblk = tid >> 5;       // 0..7
    const int sblk = tid & 31;       // 0..31
    // B staging: one full t-row (32 d values) per thread
    const int trow   = tid;          // 0..255
    const int bntile = trow >> 4;
    const int bcol   = trow & 15;
    unsigned short* const bdst0 = &ldsB[0][(bntile * 32 + bcol) * 16];      // k 0..15
    unsigned short* const bdst1 = &ldsB[0][(bntile * 32 + bcol + 16) * 16]; // k 16..31
    const size_t ldsBufB = (size_t)(16 * 32 * 16);
    const float* const aBase = fc1w + (size_t)(dblk * 4) * IN_F + s0 + sblk * 4;
    const float* const bBase = fc2w + (size_t)(t0 + trow) * HID_F;
    unsigned short* const adst = &ldsA[0][(sblk * 4) * 40 + dblk * 4];
    const size_t ldsBufA = (size_t)(128 * 40);

    f4u ar0, ar1, ar2, ar3;
    f4u br[8];

    // ---- prologue: load K-step 0 and stage to buffer 0 ----
    {
        const float* a = aBase;
        ar0 = *(const f4u*)(a);
        ar1 = *(const f4u*)(a + IN_F);
        ar2 = *(const f4u*)(a + 2 * IN_F);
        ar3 = *(const f4u*)(a + 3 * IN_F);
        #pragma unroll
        for (int c = 0; c < 8; ++c) br[c] = *(const f4u*)(bBase + 4 * c);

        u2 c0 = { pk2(ar0.x, ar1.x), pk2(ar2.x, ar3.x) };
        u2 c1 = { pk2(ar0.y, ar1.y), pk2(ar2.y, ar3.y) };
        u2 c2 = { pk2(ar0.z, ar1.z), pk2(ar2.z, ar3.z) };
        u2 c3 = { pk2(ar0.w, ar1.w), pk2(ar2.w, ar3.w) };
        *(u2*)(adst) = c0; *(u2*)(adst + 40) = c1;
        *(u2*)(adst + 80) = c2; *(u2*)(adst + 120) = c3;

        u4 q0 = { pk2(br[0].x, br[0].y), pk2(br[0].z, br[0].w), pk2(br[1].x, br[1].y), pk2(br[1].z, br[1].w) };
        u4 q1 = { pk2(br[2].x, br[2].y), pk2(br[2].z, br[2].w), pk2(br[3].x, br[3].y), pk2(br[3].z, br[3].w) };
        u4 q2 = { pk2(br[4].x, br[4].y), pk2(br[4].z, br[4].w), pk2(br[5].x, br[5].y), pk2(br[5].z, br[5].w) };
        u4 q3 = { pk2(br[6].x, br[6].y), pk2(br[6].z, br[6].w), pk2(br[7].x, br[7].y), pk2(br[7].z, br[7].w) };
        *(u4*)(bdst0) = q0; *(u4*)(bdst0 + 8) = q1;
        *(u4*)(bdst1) = q2; *(u4*)(bdst1 + 8) = q3;
    }

    for (int kb = 0; kb < KT1; ++kb) {
        __syncthreads();                    // buffer kb&1 ready for all waves
        const int cur = kb & 1;
        const bool more = (kb + 1) < KT1;

        // ---- issue next K-step global loads early (overlap with WMMA) ----
        if (more) {
            const int d0 = (kb + 1) * 32;
            const float* a = aBase + (size_t)d0 * IN_F;
            ar0 = *(const f4u*)(a);
            ar1 = *(const f4u*)(a + IN_F);
            ar2 = *(const f4u*)(a + 2 * IN_F);
            ar3 = *(const f4u*)(a + 3 * IN_F);
            #pragma unroll
            for (int c = 0; c < 8; ++c) br[c] = *(const f4u*)(bBase + d0 + 4 * c);
        }

        // ---- fragments + WMMA on current buffer ----
        FragU afr[4];
        #pragma unroll
        for (int mt = 0; mt < 4; ++mt) {
            const int row  = wm * 64 + mt * 16 + (lane & 15);
            const int koff = (lane >> 4) * 16;
            const unsigned short* p = &ldsA[cur][row * 40 + koff];
            afr[mt].q[0] = *(const u4*)(p);
            afr[mt].q[1] = *(const u4*)(p + 8);
        }
        FragU bfr[4];
        #pragma unroll
        for (int nt = 0; nt < 4; ++nt) {
            const unsigned short* p = &ldsB[cur][((wn * 4 + nt) * 32 + lane) * 16];
            bfr[nt].q[0] = *(const u4*)(p);
            bfr[nt].q[1] = *(const u4*)(p + 8);
        }
        #pragma unroll
        for (int mt = 0; mt < 4; ++mt)
            #pragma unroll
            for (int nt = 0; nt < 4; ++nt)
                acc[mt][nt].v = __builtin_amdgcn_wmma_f32_16x16x32_bf16(
                    false, afr[mt].v, false, bfr[nt].v,
                    (short)0, acc[mt][nt].v, false, false);

        // ---- stage next K-step into alternate buffer (no extra barrier) ----
        if (more) {
            const size_t off = (size_t)(cur ^ 1);
            unsigned short* ad = adst + off * ldsBufA;
            u2 c0 = { pk2(ar0.x, ar1.x), pk2(ar2.x, ar3.x) };
            u2 c1 = { pk2(ar0.y, ar1.y), pk2(ar2.y, ar3.y) };
            u2 c2 = { pk2(ar0.z, ar1.z), pk2(ar2.z, ar3.z) };
            u2 c3 = { pk2(ar0.w, ar1.w), pk2(ar2.w, ar3.w) };
            *(u2*)(ad) = c0; *(u2*)(ad + 40) = c1;
            *(u2*)(ad + 80) = c2; *(u2*)(ad + 120) = c3;

            unsigned short* b0 = bdst0 + off * ldsBufB;
            unsigned short* b1 = bdst1 + off * ldsBufB;
            u4 q0 = { pk2(br[0].x, br[0].y), pk2(br[0].z, br[0].w), pk2(br[1].x, br[1].y), pk2(br[1].z, br[1].w) };
            u4 q1 = { pk2(br[2].x, br[2].y), pk2(br[2].z, br[2].w), pk2(br[3].x, br[3].y), pk2(br[3].z, br[3].w) };
            u4 q2 = { pk2(br[4].x, br[4].y), pk2(br[4].z, br[4].w), pk2(br[5].x, br[5].y), pk2(br[5].z, br[5].w) };
            u4 q3 = { pk2(br[6].x, br[6].y), pk2(br[6].z, br[6].w), pk2(br[7].x, br[7].y), pk2(br[7].z, br[7].w) };
            *(u4*)(b0) = q0; *(u4*)(b0 + 8) = q1;
            *(u4*)(b1) = q2; *(u4*)(b1 + 8) = q3;
        }
    }

    // ---- epilogue: +bias, *mask, bf16, swizzled store ----
    #pragma unroll
    for (int mt = 0; mt < 4; ++mt) {
        #pragma unroll
        for (int nt = 0; nt < 4; ++nt) {
            const int sbase = s0 + wm * 64 + mt * 16 + ((lane >> 4) << 3);
            const int t     = t0 + wn * 64 + nt * 16 + (lane & 15);
            const float bias = fc2b[t];
            #pragma unroll
            for (int r = 0; r < 8; ++r) {
                const int s = sbase + r;
                const float v = (acc[mt][nt].f[r] + bias) * mask[(size_t)s * OUT_F + t];
                const int ntg    = t >> 4;
                const int ktg    = s >> 5;
                const int lanesw = (t & 15) + (((s >> 4) & 1) << 4);
                const int elem   = s & 15;
                bsw[((size_t)(ntg * KT2 + ktg) * 32 + lanesw) * 16 + elem] =
                    (unsigned short)f2bf1(v);
            }
        }
    }
}

// =====================================================================
// Kernel 2: out[m,t] = gelu( sum_s x[m,s] * attn[s,t] ),  exact erf GELU
// WG tile 128(m) x 256(n); wave tile 64x64; B from L2-resident swizzled buf.
// =====================================================================
__global__ __launch_bounds__(256)
void gemm2_gelu(const float* __restrict__ x,            // [M2][IN_F]
                const unsigned short* __restrict__ bsw, // swizzled bf16 attn
                float* __restrict__ out)                // [M2][OUT_F]
{
    __shared__ alignas(16) unsigned short ldsA[2][128 * 40];   // [m][k] stride 40

    const int tid  = threadIdx.x;
    const int lane = tid & 31;
    const int wave = tid >> 5;
    const int wm   = wave >> 2;      // 0..1
    const int wn   = wave & 3;       // 0..3
    const int m0   = blockIdx.y * 128;
    const int n0   = blockIdx.x * 256;

    AccU acc[4][4];
    #pragma unroll
    for (int i = 0; i < 4; ++i)
        #pragma unroll
        for (int j = 0; j < 4; ++j)
            #pragma unroll
            for (int r = 0; r < 8; ++r) acc[i][j].f[r] = 0.0f;

    const int srow  = tid >> 1;      // 0..127
    const int shalf = tid & 1;       // k half: 0 / 16
    const float* const xrow = x + (size_t)(m0 + srow) * IN_F + shalf * 16;
    unsigned short* const adst = &ldsA[0][srow * 40 + shalf * 16];
    const size_t ldsBufA = (size_t)(128 * 40);
    const int ntg0 = (n0 + wn * 64) >> 4;   // first global n-tile of this wave

    f4u f0, f1, f2, f3;

    // ---- prologue: stage K-step 0 ----
    {
        const float* src = xrow;
        f0 = *(const f4u*)(src);
        f1 = *(const f4u*)(src + 4);
        f2 = *(const f4u*)(src + 8);
        f3 = *(const f4u*)(src + 12);
        u4 lo = { pk2(f0.x, f0.y), pk2(f0.z, f0.w), pk2(f1.x, f1.y), pk2(f1.z, f1.w) };
        u4 hi = { pk2(f2.x, f2.y), pk2(f2.z, f2.w), pk2(f3.x, f3.y), pk2(f3.z, f3.w) };
        *(u4*)(adst) = lo; *(u4*)(adst + 8) = hi;
    }

    for (int kb = 0; kb < KT2; ++kb) {
        __syncthreads();
        const int cur = kb & 1;
        const bool more = (kb + 1) < KT2;

        if (more) {
            const float* src = xrow + (kb + 1) * 32;
            f0 = *(const f4u*)(src);
            f1 = *(const f4u*)(src + 4);
            f2 = *(const f4u*)(src + 8);
            f3 = *(const f4u*)(src + 12);
            if (kb + 2 < KT2) __builtin_prefetch(src + 32, 0, 3);
        }

        FragU afr[4];
        #pragma unroll
        for (int mt = 0; mt < 4; ++mt) {
            const int row  = wm * 64 + mt * 16 + (lane & 15);
            const int koff = (lane >> 4) * 16;
            const unsigned short* p = &ldsA[cur][row * 40 + koff];
            afr[mt].q[0] = *(const u4*)(p);
            afr[mt].q[1] = *(const u4*)(p + 8);
        }
        FragU bfr[4];
        #pragma unroll
        for (int nt = 0; nt < 4; ++nt) {
            const unsigned short* p =
                bsw + ((size_t)((ntg0 + nt) * KT2 + kb) * 32 + lane) * 16;
            bfr[nt].q[0] = *(const u4*)(p);      // L2-resident (2 MB total)
            bfr[nt].q[1] = *(const u4*)(p + 8);
        }
        #pragma unroll
        for (int mt = 0; mt < 4; ++mt)
            #pragma unroll
            for (int nt = 0; nt < 4; ++nt)
                acc[mt][nt].v = __builtin_amdgcn_wmma_f32_16x16x32_bf16(
                    false, afr[mt].v, false, bfr[nt].v,
                    (short)0, acc[mt][nt].v, false, false);

        if (more) {
            unsigned short* ad = adst + (size_t)(cur ^ 1) * ldsBufA;
            u4 lo = { pk2(f0.x, f0.y), pk2(f0.z, f0.w), pk2(f1.x, f1.y), pk2(f1.z, f1.w) };
            u4 hi = { pk2(f2.x, f2.y), pk2(f2.z, f2.w), pk2(f3.x, f3.y), pk2(f3.z, f3.w) };
            *(u4*)(ad) = lo; *(u4*)(ad + 8) = hi;
        }
    }

    // ---- epilogue: exact GELU, f32 store ----
    #pragma unroll
    for (int mt = 0; mt < 4; ++mt) {
        #pragma unroll
        for (int nt = 0; nt < 4; ++nt) {
            const int gmb = m0 + wm * 64 + mt * 16 + ((lane >> 4) << 3);
            const int gn  = n0 + wn * 64 + nt * 16 + (lane & 15);
            #pragma unroll
            for (int r = 0; r < 8; ++r) {
                const float v = acc[mt][nt].f[r];
                const float g = 0.5f * v * (1.0f + erff(v * 0.70710678118654752f));
                out[(size_t)(gmb + r) * OUT_F + gn] = g;
            }
        }
    }
}

// =====================================================================
extern "C" void kernel_launch(void* const* d_in, const int* in_sizes, int n_in,
                              void* d_out, int out_size, void* d_ws, size_t ws_size,
                              hipStream_t stream) {
    const float* x     = (const float*)d_in[0];
    const float* fc1w  = (const float*)d_in[1];
    const float* fc2w  = (const float*)d_in[2];
    const float* fc2b  = (const float*)d_in[3];
    const float* mask  = (const float*)d_in[4];
    unsigned short* bsw = (unsigned short*)d_ws;     // 1024*1024 bf16 = 2 MB

    gemm1_bias_mask<<<dim3(OUT_F / 256, IN_F / 128), 256, 0, stream>>>(
        fc1w, fc2w, fc2b, mask, bsw);
    gemm2_gelu<<<dim3(OUT_F / 256, M2 / 128), 256, 0, stream>>>(
        x, bsw, (float*)d_out);
}
